// FeatureAdaptionCls_5291399708999
// MI455X (gfx1250) — compile-verified, matrix operand order checked by reading, not run on hardware
//
#include <hip/hip_runtime.h>
#include <hip/hip_bf16.h>

// Problem constants (from reference)
#define BATCH 4
#define CIN   256
#define COUT  256
#define HH    64
#define WW    64
#define HW    (HH*WW)          // 4096
#define DG    4
#define CG    (CIN/DG)         // 64
#define KK    9
#define KDIM  (CIN*KK)         // 2304
#define NSAMP (DG*KK)          // 36 distinct sample maps per batch
#define OFFC  (DG*2*KK)        // 72

// Main GEMM tiling
#define NT    64               // pixels per block (N tile)
#define CCH   32               // channels per K chunk
#define KCH   (CCH*KK)         // 288 K per chunk = 9 WMMA K-steps
#define NCHUNK (CIN/CCH)       // 8
#define BK    (KCH + 8)        // padded LDS stride (296 halves = 592B, 16B aligned)

typedef __attribute__((ext_vector_type(16))) _Float16 v16h;
typedef __attribute__((ext_vector_type(8)))  _Float16 v8h;
typedef __attribute__((ext_vector_type(8)))  float    v8f;

// Build a 16-half WMMA fragment from two contiguous 8-half groups
// (halves 0..7 at base, halves 8..15 at base+16) -- CDNA5 16-bit A layout:
// per-lane K {0..7 | 8..15} then {16..23 | 24..31}.
template <typename PT>
static __device__ inline v16h make_frag(PT base) {
    v8h lo = *(const v8h*)(base);
    v8h hi = *(const v8h*)(base + 16);
    return __builtin_shufflevector(lo, hi, 0,1,2,3,4,5,6,7,8,9,10,11,12,13,14,15);
}

// ---------------------------------------------------------------------------
// Kernel 1: flat fp32 -> f16 weight conversion (layout already (COUT, c*9+kk)).
// ---------------------------------------------------------------------------
__global__ void wconv_f16_kernel(const float* __restrict__ w, _Float16* __restrict__ w16) {
    int i = blockIdx.x * 256 + threadIdx.x;
    if (i < COUT * KDIM) w16[i] = (_Float16)w[i];
}

// ---------------------------------------------------------------------------
// Kernel 2: fused offset-conv + bilinear metadata precompute.
// One thread per (b, s=dg*9+kk, p). Computes dy,dx via the 2-in 3x3 conv,
// then packs clamped corner indices (idx00 | idx11<<12) and validity-masked
// bilinear weights (float4). Shared by all 64 channels of the deform group.
// ---------------------------------------------------------------------------
__global__ void prep_kernel(const float* __restrict__ shp,
                            const float* __restrict__ wofs,
                            unsigned int* __restrict__ idxpk,
                            float4* __restrict__ wts) {
    int idx = blockIdx.x * 256 + threadIdx.x;
    if (idx >= BATCH * NSAMP * HW) return;
    int p  = idx & (HW - 1);
    int s  = (idx >> 12) % NSAMP;      // dg*9 + kk
    int b  = idx / (NSAMP * HW);
    int kk = s % KK;
    int py = p >> 6, px = p & 63;

    // offset conv: output channels 2s (dy), 2s+1 (dx)
    float dy = 0.f, dx = 0.f;
    #pragma unroll
    for (int ic = 0; ic < 2; ++ic) {
        const float* sb  = shp + (size_t)(b * 2 + ic) * HW;
        const float* wby = wofs + (size_t)((2 * s) * 2 + ic) * KK;
        const float* wbx = wofs + (size_t)((2 * s + 1) * 2 + ic) * KK;
        #pragma unroll
        for (int ky = 0; ky < 3; ++ky) {
            int yy = py + ky - 1;
            if (yy < 0 || yy >= HH) continue;
            #pragma unroll
            for (int kx = 0; kx < 3; ++kx) {
                int xx = px + kx - 1;
                if (xx < 0 || xx >= WW) continue;
                float sv = sb[yy * WW + xx];
                dy += sv * wby[ky * 3 + kx];
                dx += sv * wbx[ky * 3 + kx];
            }
        }
    }

    // bilinear sample metadata
    float sy = dy + (float)(py + (kk / 3) - 1);
    float sx = dx + (float)(px + (kk % 3) - 1);
    float y0f = floorf(sy), x0f = floorf(sx);
    float ly = sy - y0f, lx = sx - x0f;
    int y0 = (int)y0f, x0 = (int)x0f;

    bool y0in = (y0 >= 0) & (y0 < HH);
    bool y1in = (y0 >= -1) & (y0 < HH - 1);
    bool x0in = (x0 >= 0) & (x0 < WW);
    bool x1in = (x0 >= -1) & (x0 < WW - 1);

    float w00 = (y0in & x0in) ? (1.f - ly) * (1.f - lx) : 0.f;
    float w01 = (y0in & x1in) ? (1.f - ly) * lx         : 0.f;
    float w10 = (y1in & x0in) ? ly * (1.f - lx)         : 0.f;
    float w11 = (y1in & x1in) ? ly * lx                 : 0.f;

    int yc0 = min(max(y0, 0), HH - 1);
    int xc0 = min(max(x0, 0), WW - 1);
    int yc1 = min(max(y0 + 1, 0), HH - 1);
    int xc1 = min(max(x0 + 1, 0), WW - 1);
    unsigned int i00 = (unsigned int)(yc0 * WW + xc0);
    unsigned int i11 = (unsigned int)(yc1 * WW + xc1);

    idxpk[idx] = i00 | (i11 << 12);
    wts[idx]   = make_float4(w00, w01, w10, w11);
}

// ---------------------------------------------------------------------------
// Kernel 3: fused gather-im2col + WMMA GEMM + ReLU.
// Block = 256 threads (8 wave32). Each block: full M=256 x 64-pixel N tile for
// one batch. K chunks of 288 (32 channels x 9 taps): metadata loaded once per
// tap, reused across 8 channels per thread; then 9 WMMA K-steps per chunk.
// ---------------------------------------------------------------------------
__global__ void __launch_bounds__(256)
dcn_wmma_kernel(const float* __restrict__ x,
                const unsigned int* __restrict__ idxpk,
                const float4* __restrict__ wts,
                const _Float16* __restrict__ w16,
                float* __restrict__ out) {
    __shared__ _Float16 Bs[NT * BK];   // B tile, [n][k_local], padded stride

    const int b     = blockIdx.x >> 6;
    const int pbase = (blockIdx.x & 63) * NT;
    const int tid   = threadIdx.x;
    const int lane  = tid & 31;
    const int lhalf = lane >> 4;
    const int l16   = lane & 15;
    const int mbase = (tid >> 5) * 32;     // wave's 32-row M strip
    const int koffs = lhalf << 3;          // fragment K sub-offset (halves)

    const int fn  = tid & 63;              // pixel within tile (coalesced)
    const int fk0 = tid >> 6;              // 0..3 -> 8-channel slice of chunk
    const int p   = pbase + fn;

    const float* xb_batch = x + (size_t)b * CIN * HW;
    const unsigned int* ib_batch = idxpk + ((size_t)b * NSAMP << 12) + p;
    const float4*       wb_batch = wts   + ((size_t)b * NSAMP << 12) + p;

    v8f acc[2][4] = {};

    for (int chunk = 0; chunk < NCHUNK; ++chunk) {
        const int cbase = chunk * CCH;
        const int dg    = cbase >> 6;

        if (chunk) __syncthreads();        // WMMA reads of previous chunk done

        // ---- gather fill: 288 x 64 B tile, 72 samples/thread ----
        {
            const unsigned int* ib = ib_batch + ((size_t)(dg * KK) << 12);
            const float4*       wb = wb_batch + ((size_t)(dg * KK) << 12);
            const float* xcb = xb_batch + (size_t)(cbase + fk0 * 8) * HW;
            _Float16* brow = &Bs[fn * BK + fk0 * 8 * KK];
            #pragma unroll
            for (int kk = 0; kk < KK; ++kk) {
                unsigned int pk = ib[(size_t)kk << 12];
                float4 wv = wb[(size_t)kk << 12];
                int i00 = pk & 4095;
                int i11 = (pk >> 12) & 4095;
                int i01 = (i00 & 0xFC0) | (i11 & 63);
                int i10 = (i11 & 0xFC0) | (i00 & 63);
                #pragma unroll
                for (int cc = 0; cc < 8; ++cc) {
                    const float* xp = xcb + (size_t)cc * HW;
                    float v = xp[i00] * wv.x + xp[i01] * wv.y
                            + xp[i10] * wv.z + xp[i11] * wv.w;
                    brow[cc * KK + kk] = (_Float16)v;
                }
            }
        }
        __syncthreads();                   // fill visible to all waves

        // ---- 9 WMMA K-steps over this chunk ----
        const int kbase = cbase * KK;      // chunk * 288
        for (int ks = 0; ks < KCH / 32; ++ks) {
            const _Float16* wp =
                w16 + (size_t)(mbase + l16) * KDIM + kbase + ks * 32 + koffs;
            v16h a0 = make_frag(wp);
            v16h a1 = make_frag(wp + (size_t)16 * KDIM);
            #pragma unroll
            for (int ni = 0; ni < 4; ++ni) {
                v16h bf = make_frag(&Bs[(ni * 16 + l16) * BK + ks * 32 + koffs]);
                acc[0][ni] = __builtin_amdgcn_wmma_f32_16x16x32_f16(
                    false, a0, false, bf, (short)0, acc[0][ni], false, false);
                acc[1][ni] = __builtin_amdgcn_wmma_f32_16x16x32_f16(
                    false, a1, false, bf, (short)0, acc[1][ni], false, false);
            }
        }
    }

    // ---- ReLU + store. C/D layout: vgpr r, lanes 0-15 -> M=r, N=lane;
    //      lanes 16-31 -> M=r+8, N=lane-16. ----
    float* ob = out + (size_t)b * COUT * HW + pbase;
    #pragma unroll
    for (int mi = 0; mi < 2; ++mi) {
        #pragma unroll
        for (int r = 0; r < 8; ++r) {
            int m = mbase + mi * 16 + r + (lhalf << 3);
            float* orow = ob + (size_t)m * HW + l16;
            #pragma unroll
            for (int ni = 0; ni < 4; ++ni) {
                float v = acc[mi][ni][r];
                orow[ni * 16] = v > 0.f ? v : 0.f;
            }
        }
    }
}

// ---------------------------------------------------------------------------
extern "C" void kernel_launch(void* const* d_in, const int* in_sizes, int n_in,
                              void* d_out, int out_size, void* d_ws, size_t ws_size,
                              hipStream_t stream) {
    const float* x        = (const float*)d_in[0];
    const float* shp      = (const float*)d_in[1];
    const float* wofs     = (const float*)d_in[2];
    const float* w_deform = (const float*)d_in[3];
    float* out            = (float*)d_out;

    // workspace layout (16B aligned regions):
    //   wts:   4*36*4096 * float4 = 9,437,184 B
    //   idxpk: 4*36*4096 * u32    = 2,359,296 B
    //   w16:   256*2304  * f16    = 1,179,648 B
    float4*       wts   = (float4*)d_ws;
    unsigned int* idxpk = (unsigned int*)((char*)d_ws + 9437184);
    _Float16*     w16   = (_Float16*)((char*)d_ws + 9437184 + 2359296);

    wconv_f16_kernel<<<(COUT * KDIM + 255) / 256, 256, 0, stream>>>(w_deform, w16);
    prep_kernel<<<(BATCH * NSAMP * HW + 255) / 256, 256, 0, stream>>>(shp, wofs, idxpk, wts);
    dcn_wmma_kernel<<<BATCH * (HW / NT), 256, 0, stream>>>(x, idxpk, wts, w16, out);
}
